// MultiHeadMaskedAttention_4904852652327
// MI455X (gfx1250) — matrix operand order repaired and checked by reading, not compiled, not run on hardware
//
#include <hip/hip_runtime.h>

// ---------------- CDNA5 (gfx1250) types & WMMA helpers ----------------
typedef __attribute__((ext_vector_type(16))) __bf16 v16bf;
typedef __attribute__((ext_vector_type(8)))  __bf16 v8bf;
typedef __attribute__((ext_vector_type(8)))  float  v8f;

#define S_LEN   2048
#define DMODEL  1024
#define NHEAD   16
#define HDIM    64

// D = A(16x32 bf16) * B(32x16 bf16) + C(16x16 f32)
__device__ __forceinline__ v8f wmma_bf16(v16bf a, v16bf b, v8f c) {
  return __builtin_amdgcn_wmma_f32_16x16x32_bf16(
      /*neg_a=*/false, a, /*neg_b=*/false, b,
      /*c_mod=*/(short)0, c, /*reuse_a=*/false, /*reuse_b=*/false);
}

// A-fragment (16x32, 16-bit): lane l (row M=l&15) holds the K-contiguous
// run [h*16, h*16+16) where h = l>>4 (matches B-fragment order).
__device__ __forceinline__ v16bf ldfrag_a_bf16(const __bf16* row, int half) {
  v8bf lo = *(const v8bf*)(row + half * 16);
  v8bf hi = *(const v8bf*)(row + half * 16 + 8);
  v16bf r;
#pragma unroll
  for (int i = 0; i < 8; ++i) { r[i] = lo[i]; r[i + 8] = hi[i]; }
  return r;
}
// B-fragment (32x16, 16-bit): lane l = column N=l&15, K-contiguous run of 16
// starting at (l>>4)*16 (caller pre-applies the k offset).
__device__ __forceinline__ v16bf ldfrag_b_bf16(const __bf16* p) {
  return *(const v16bf*)p;
}

// ---------------- One-shot fp32 -> bf16 conversion (8 elems/thread) -------
__global__ void cvt_f32_to_bf16(const float* __restrict__ in,
                                unsigned short* __restrict__ out, int n8) {
  const int i = blockIdx.x * blockDim.x + threadIdx.x;
  if (i >= n8) return;
  const float4* p = (const float4*)in + (size_t)i * 2;
  const float4 a = p[0], b = p[1];
  v8bf r;
  r[0] = (__bf16)a.x; r[1] = (__bf16)a.y; r[2] = (__bf16)a.z; r[3] = (__bf16)a.w;
  r[4] = (__bf16)b.x; r[5] = (__bf16)b.y; r[6] = (__bf16)b.z; r[7] = (__bf16)b.w;
  *((v8bf*)out + i) = r;
}

// ---------------- Projection GEMM: Y = X @ W^T + b (all-bf16 operands) ----
// mode 0: Y -> bf16 head layout [B,H,S,Dh]      (Q, K)
// mode 1: Y -> bf16 transposed  [B,H,Dh,S]      (V)
// mode 2: Y -> f32 flat [B*S, D]                (final output)
// Each wave computes a 32(m) x 64(n) tile: 2 A-frags x 4 B-frags -> 8 WMMAs
// per 32-deep k-step; each B-frag is reused across both row groups.
__global__ void proj_gemm(const unsigned short* __restrict__ Abf_us,
                          const unsigned short* __restrict__ Wbf_us,
                          const float* __restrict__ bias,
                          unsigned short* __restrict__ out_us,
                          float* __restrict__ out_f, int mode) {
  const int lane = threadIdx.x & 31;
  const int wave = threadIdx.x >> 5;
  const int lr = lane & 15, half = lane >> 4;
  const int mt = blockIdx.y * 8 + wave;      // 32-row tile, 0..255
  const int n0 = blockIdx.x * 64;            // 64-col strip
  const int m0 = mt * 32;
  const int K = DMODEL;
  const __bf16* Ab = (const __bf16*)Abf_us;
  const __bf16* Wb = (const __bf16*)Wbf_us;
  __bf16* ob = (__bf16*)out_us;

  v8f acc[2][4] = {};
  const int arow0 = (m0 + lr) * K;
  const int arow1 = (m0 + 16 + lr) * K;
#pragma unroll 4
  for (int k0 = 0; k0 < K; k0 += 32) {
    const v16bf a0 = ldfrag_a_bf16(Ab + arow0 + k0, half);
    const v16bf a1 = ldfrag_a_bf16(Ab + arow1 + k0, half);
#pragma unroll
    for (int j = 0; j < 4; ++j) {
      const int n = n0 + j * 16 + lr;        // lane's B column
      const v16bf b = ldfrag_b_bf16(Wb + n * K + k0 + half * 16);
      acc[0][j] = wmma_bf16(a0, b, acc[0][j]);
      acc[1][j] = wmma_bf16(a1, b, acc[1][j]);
    }
  }
  // Epilogue: D layout -> lane holds col n, rows m = r + 8*half (r = 0..7).
#pragma unroll
  for (int g = 0; g < 2; ++g) {
    const int mg = m0 + g * 16;
#pragma unroll
    for (int j = 0; j < 4; ++j) {
      const int n = n0 + j * 16 + lr;
      const float bv = bias[n];
#pragma unroll
      for (int r = 0; r < 8; ++r) {
        const int m = mg + r + 8 * half;
        const float val = acc[g][j][r] + bv;
        if (mode == 2) {
          out_f[m * DMODEL + n] = val;
        } else {
          const int b_ = m >> 11, s = m & (S_LEN - 1);
          const int h = n >> 6, d = n & (HDIM - 1);
          const int idx = (mode == 0)
              ? (((b_ * NHEAD + h) * S_LEN + s) * HDIM + d)   // [B,H,S,Dh]
              : (((b_ * NHEAD + h) * HDIM + d) * S_LEN + s);  // [B,H,Dh,S]
          ob[idx] = (__bf16)val;
        }
      }
    }
  }
}

// ---------------- Column softmax stats (softmax over QUERY axis) ----------
// For each (b,h) and key column k: m[k] = max_q s[q,k], l[k] = sum_q exp(s-m).
// D-fragment layout puts column k's scores in lanes {k%16, k%16+16} -> pure
// per-lane online accumulation + one shfl_xor(16) combine.
__global__ void attn_stats(const unsigned short* __restrict__ Qh_us,
                           const unsigned short* __restrict__ Kh_us,
                           float* __restrict__ colmax,
                           float* __restrict__ colsum) {
  const int lane = threadIdx.x & 31, wave = threadIdx.x >> 5;
  const int lr = lane & 15, half = lane >> 4;
  const int bh = blockIdx.x;                  // 0..63  (b*H + h)
  const int kt = blockIdx.y * 8 + wave;       // column tile 0..127
  const __bf16* Qb = (const __bf16*)Qh_us + bh * S_LEN * HDIM;
  const __bf16* Kb = (const __bf16*)Kh_us + bh * S_LEN * HDIM;

  const int kcol = kt * 16 + lr;              // lane's key column
  const v16bf bk0 = ldfrag_b_bf16(Kb + kcol * HDIM + 0 + half * 16);
  const v16bf bk1 = ldfrag_b_bf16(Kb + kcol * HDIM + 32 + half * 16);

  float mrun = -1e30f, srun = 0.0f;
  for (int qt = 0; qt < S_LEN / 16; ++qt) {
    const int qrow = qt * 16 + lr;
    v16bf a0 = ldfrag_a_bf16(Qb + qrow * HDIM + 0, half);
    v16bf a1 = ldfrag_a_bf16(Qb + qrow * HDIM + 32, half);
    v8f d = {};
    d = wmma_bf16(a0, bk0, d);
    d = wmma_bf16(a1, bk1, d);
    float tmax = d[0];
#pragma unroll
    for (int r = 1; r < 8; ++r) tmax = fmaxf(tmax, d[r]);
    const float nm = fmaxf(mrun, tmax);
    float ts = 0.0f;
#pragma unroll
    for (int r = 0; r < 8; ++r) ts += __expf(d[r] - nm);
    srun = srun * __expf(mrun - nm) + ts;
    mrun = nm;
  }
  // combine the two q-row halves held by lane pair (l, l^16)
  const float mo = __shfl_xor(mrun, 16, 32);
  const float so = __shfl_xor(srun, 16, 32);
  const float mf = fmaxf(mrun, mo);
  const float sf = srun * __expf(mrun - mf) + so * __expf(mo - mf);
  if (half == 0) {
    colmax[bh * S_LEN + kcol] = mf;
    colsum[bh * S_LEN + kcol] = sf;
  }
}

// ---------------- Apply: out[q,:] = sum_k exp(s-m)/l * (1/8) * V[k,:] ------
__global__ void attn_apply(const unsigned short* __restrict__ Qh_us,
                           const unsigned short* __restrict__ Kh_us,
                           const unsigned short* __restrict__ Vt_us,
                           const float* __restrict__ colmax,
                           const float* __restrict__ colsum,
                           unsigned short* __restrict__ O_us) {
  __shared__ __bf16 pstage[8][16 * 32];       // per-wave 16q x 32k prob tile
  const int lane = threadIdx.x & 31, wave = threadIdx.x >> 5;
  const int lr = lane & 15, half = lane >> 4;
  const int bh = blockIdx.x;
  const int b_ = bh >> 4, h = bh & (NHEAD - 1);
  const int qt = blockIdx.y * 8 + wave;       // query tile 0..127
  const __bf16* Qb = (const __bf16*)Qh_us + bh * S_LEN * HDIM;
  const __bf16* Kb = (const __bf16*)Kh_us + bh * S_LEN * HDIM;
  const __bf16* Vb = (const __bf16*)Vt_us + bh * HDIM * S_LEN;  // [Dh,S]
  __bf16* O = (__bf16*)O_us;
  const float* cm = colmax + bh * S_LEN;
  const float* cs = colsum + bh * S_LEN;
  __bf16* myp = &pstage[wave][0];

  const int qrow = qt * 16 + lr;
  const v16bf aq0 = ldfrag_a_bf16(Qb + qrow * HDIM + 0, half);
  const v16bf aq1 = ldfrag_a_bf16(Qb + qrow * HDIM + 32, half);

  v8f acc[4] = {};
  for (int k0 = 0; k0 < S_LEN; k0 += 32) {
#pragma unroll
    for (int sub = 0; sub < 2; ++sub) {
      const int kcol = k0 + sub * 16 + lr;
      v16bf bk0 = ldfrag_b_bf16(Kb + kcol * HDIM + 0 + half * 16);
      v16bf bk1 = ldfrag_b_bf16(Kb + kcol * HDIM + 32 + half * 16);
      v8f d = {};
      d = wmma_bf16(aq0, bk0, d);
      d = wmma_bf16(aq1, bk1, d);
      const float mv = cm[kcol];
      const float rl = 0.125f / cs[kcol];     // 1/sqrt(Dh) applied post-softmax
#pragma unroll
      for (int r = 0; r < 8; ++r) {           // D row = r + 8*half, col = lane
        const float p = __expf(d[r] - mv) * rl;
        myp[(r + 8 * half) * 32 + sub * 16 + lr] = (__bf16)p;
      }
    }
    // cross-lane LDS visibility inside the wave (CDNA5 split counters)
    asm volatile("s_wait_dscnt 0" ::: "memory");
    // Re-read as A fragment: row lr, K-contiguous run half*16..half*16+15
    v8bf lo = *(const v8bf*)(myp + lr * 32 + half * 16);
    v8bf hi = *(const v8bf*)(myp + lr * 32 + half * 16 + 8);
    v16bf ap;
#pragma unroll
    for (int i = 0; i < 8; ++i) { ap[i] = lo[i]; ap[i + 8] = hi[i]; }
#pragma unroll
    for (int j = 0; j < 4; ++j) {             // 4 x 16 output (Dh=64) columns
      const int dcol = j * 16 + lr;
      v16bf bv = ldfrag_b_bf16(Vb + dcol * S_LEN + k0 + half * 16);
      acc[j] = wmma_bf16(ap, bv, acc[j]);
    }
  }
  // Store attention output to bf16 [B,S,D] (D index = h*64 + d)
#pragma unroll
  for (int j = 0; j < 4; ++j) {
#pragma unroll
    for (int r = 0; r < 8; ++r) {
      const int q = qt * 16 + r + 8 * half;
      const int n = h * HDIM + j * 16 + lr;
      O[(b_ * S_LEN + q) * DMODEL + n] = (__bf16)acc[j][r];
    }
  }
}

// ---------------- Host-side orchestration ----------------
extern "C" void kernel_launch(void* const* d_in, const int* in_sizes, int n_in,
                              void* d_out, int out_size, void* d_ws, size_t ws_size,
                              hipStream_t stream) {
  (void)in_sizes; (void)n_in; (void)out_size; (void)ws_size;
  const float* queries = (const float*)d_in[0];
  const float* keys    = (const float*)d_in[1];
  const float* values  = (const float*)d_in[2];
  const float* Wq_w = (const float*)d_in[3];
  const float* Wq_b = (const float*)d_in[4];
  const float* Wk_w = (const float*)d_in[5];
  const float* Wk_b = (const float*)d_in[6];
  const float* Wv_w = (const float*)d_in[7];
  const float* Wv_b = (const float*)d_in[8];
  const float* Wo_w = (const float*)d_in[9];
  const float* Wo_b = (const float*)d_in[10];

  char* ws = (char*)d_ws;                       // ~83 MB used
  unsigned short* Qh  = (unsigned short*)(ws);                      // 16 MB
  unsigned short* Kh  = (unsigned short*)(ws + (16u << 20));        // 16 MB
  unsigned short* Vt  = (unsigned short*)(ws + (32u << 20));        // 16 MB
  float*          cm  = (float*)(ws + (48u << 20));                 // 512 KB
  float*          cs  = (float*)(ws + (48u << 20) + (512u << 10));  // 512 KB
  unsigned short* O   = (unsigned short*)(ws + (49u << 20));        // 16 MB
  unsigned short* Xbf = (unsigned short*)(ws + (65u << 20));        // 16 MB staging
  unsigned short* Wbf = (unsigned short*)(ws + (81u << 20));        // 2 MB staging

  const int nX8 = (4 * S_LEN * DMODEL) / 8;     // 1M vec8 chunks
  const int nW8 = (DMODEL * DMODEL) / 8;        // 128K vec8 chunks
  const dim3 gX(nX8 / 256), gW(nW8 / 256);
  const dim3 gProj(DMODEL / 64, (4 * S_LEN) / (32 * 8));  // (16, 32)
  const dim3 gAttn(4 * NHEAD, (S_LEN / 16) / 8);          // (64, 16)

  // Q projection
  cvt_f32_to_bf16<<<gX, 256, 0, stream>>>(queries, Xbf, nX8);
  cvt_f32_to_bf16<<<gW, 256, 0, stream>>>(Wq_w, Wbf, nW8);
  proj_gemm<<<gProj, 256, 0, stream>>>(Xbf, Wbf, Wq_b, Qh, nullptr, 0);
  // K projection
  cvt_f32_to_bf16<<<gX, 256, 0, stream>>>(keys, Xbf, nX8);
  cvt_f32_to_bf16<<<gW, 256, 0, stream>>>(Wk_w, Wbf, nW8);
  proj_gemm<<<gProj, 256, 0, stream>>>(Xbf, Wbf, Wk_b, Kh, nullptr, 0);
  // V projection (transposed per-head output)
  cvt_f32_to_bf16<<<gX, 256, 0, stream>>>(values, Xbf, nX8);
  cvt_f32_to_bf16<<<gW, 256, 0, stream>>>(Wv_w, Wbf, nW8);
  proj_gemm<<<gProj, 256, 0, stream>>>(Xbf, Wbf, Wv_b, Vt, nullptr, 1);
  // Attention (column softmax + apply)
  attn_stats<<<gAttn, 256, 0, stream>>>(Qh, Kh, cm, cs);
  attn_apply<<<gAttn, 256, 0, stream>>>(Qh, Kh, Vt, cm, cs, O);
  // Output projection (fp32 result straight to d_out)
  cvt_f32_to_bf16<<<gW, 256, 0, stream>>>(Wo_w, Wbf, nW8);
  proj_gemm<<<gProj, 256, 0, stream>>>(O, Wbf, Wo_b, nullptr, (float*)d_out, 2);
}
